// GAT_62182536511735
// MI455X (gfx1250) — matrix operand order repaired and checked by reading, not compiled, not run on hardware
//
#include <hip/hip_runtime.h>
#include <hip/hip_bf16.h>

// ---------------------------------------------------------------------------
// GAT (3-layer) for MI455X / gfx1250.
//  - GEMMs run on v_wmma_f32_16x16x32_bf16 (bf16 in, f32 accum), one wave per
//    16x64 output strip (4 WMMA accumulators share each A fragment).
//  - K is a template parameter (256/128): fully unrolled K-loop with all
//    fragment loads issued ahead of the WMMA chain -> staggered s_wait_loadcnt
//    instead of full drains, loads overlap the matrix pipe.
//  - Weight matrix transposed + zero-padded to a 64-col multiple: the K-loop
//    has NO exec-masked loads (WMMA wants EXEC all-1s).
//  - Edge softmax/aggregation uses L2-resident scatter atomics (the 51MB
//    feature matrix fits the 192MB L2), one wave32 per edge.
// ---------------------------------------------------------------------------

typedef __bf16 bf16;
typedef __attribute__((ext_vector_type(16))) __bf16 v16bf;
typedef __attribute__((ext_vector_type(8)))  float  v8f;
typedef __attribute__((ext_vector_type(4)))  unsigned u32x4;

union Frag {
    v16bf v;
    u32x4 q[2];
    unsigned u[8];
};

#define NEG_SLOPE 0.2f
#define NT 4                 // 16-col tiles per wave (16x64 strip)

// ---- helpers --------------------------------------------------------------

__device__ __forceinline__ unsigned fkey(float f) {
    unsigned u = __float_as_uint(f);
    return (u & 0x80000000u) ? ~u : (u | 0x80000000u);   // monotone float->uint
}
__device__ __forceinline__ float funkey(unsigned u) {
    return __uint_as_float((u & 0x80000000u) ? (u ^ 0x80000000u) : ~u);
}
__device__ __forceinline__ void edge_nodes(long eid, const int* __restrict__ esrc,
                                           const int* __restrict__ edst, long E,
                                           int& s, int& d) {
    if (eid < E) { s = esrc[eid]; d = edst[eid]; }
    else         { s = d = (int)(eid - E); }             // self loop
}

// ---- conversion kernels ---------------------------------------------------

__global__ void gat_cvt_bf16(const float* __restrict__ src,
                             unsigned short* __restrict__ dst, long n) {
    long t = (long)blockIdx.x * blockDim.x + threadIdx.x;
    if (t >= n) return;
    unsigned u = __float_as_uint(src[t]);
    unsigned r = u + 0x7fffu + ((u >> 16) & 1u);         // RNE
    dst[t] = (unsigned short)(r >> 16);
}

// W [K,F] f32 -> Wt [Fpad,K] bf16 (transposed; rows F..Fpad-1 pre-zeroed)
__global__ void gat_cvt_wt(const float* __restrict__ W,
                           unsigned short* __restrict__ Wt, int K, int F) {
    long t = (long)blockIdx.x * blockDim.x + threadIdx.x;
    if (t >= (long)K * F) return;
    int k = (int)(t / F), f = (int)(t % F);
    unsigned u = __float_as_uint(W[t]);
    unsigned r = u + 0x7fffu + ((u >> 16) & 1u);
    Wt[(size_t)f * K + k] = (unsigned short)(r >> 16);
}

__global__ void gat_zero(unsigned* __restrict__ p, long n) {
    long t = (long)blockIdx.x * blockDim.x + threadIdx.x;
    if (t < n) p[t] = 0u;
}

// ---- WMMA GEMM: Y[N,F] = X[N,K](bf16) @ Wt[Fpad,K](bf16)^T ---------------
// One wave per 16x64 output strip; K stepped by 32 via v_wmma_f32_16x16x32_bf16.
// A layout (16-bit 16x32, ISA 7.12.2): row = lane&15; lanes<16 hold K{0..7,16..23},
// lanes>=16 hold K{8..15,24..31} as consecutive pairs -> two b128 loads.
// B layout (32x16): col = lane&15; same K interleave as A (lanes<16 K{0..7,16..23}, lanes>=16 K{8..15,24..31}).
// Wt is zero-padded to 64*colBlocks rows so all loads are unconditional.

template<int K>
__global__ __launch_bounds__(256) void gat_wmma_gemm(
        const unsigned short* __restrict__ X, const unsigned short* __restrict__ Wt,
        float* __restrict__ Y, int N, int F) {
    int lane = threadIdx.x & 31;
    int wave = (int)(((long)blockIdx.x * blockDim.x + threadIdx.x) >> 5);
    int colBlocks = (F + 63) >> 6;           // groups of NT=4 col tiles
    int tilesM = N >> 4;                     // N is a multiple of 16 (100000)
    if (wave >= tilesM * colBlocks) return;
    int tm = wave / colBlocks, tb = wave % colBlocks;
    int l15 = lane & 15;
    int hi  = lane >> 4;
    int row = (tm << 4) + l15;
    int colBase = tb << 6;
    const size_t rowOff = (size_t)row * K;
    const int khalfA = hi ? 8 : 0;           // A interleave
    const int koffB  = hi ? 8 : 0;           // B interleave (matches A)

    size_t bOff[NT];
    #pragma unroll
    for (int j = 0; j < NT; ++j)
        bOff[j] = (size_t)(colBase + j * 16 + l15) * K;

    v8f acc[NT] = {};
    #pragma unroll
    for (int k0 = 0; k0 < K; k0 += 32) {
        // issue ALL loads for this K-step first, then consume: lets the
        // scheduler stagger s_wait_loadcnt and pipeline across iterations
        Frag a, b[NT];
        a.q[0] = *(const u32x4*)(X + rowOff + k0 + khalfA);
        a.q[1] = *(const u32x4*)(X + rowOff + k0 + khalfA + 16);
        #pragma unroll
        for (int j = 0; j < NT; ++j) {
            b[j].q[0] = *(const u32x4*)(Wt + bOff[j] + k0 + koffB);
            b[j].q[1] = *(const u32x4*)(Wt + bOff[j] + k0 + koffB + 16);
        }
        #pragma unroll
        for (int j = 0; j < NT; ++j)
            acc[j] = __builtin_amdgcn_wmma_f32_16x16x32_bf16(
                         false, a.v, false, b[j].v, (short)0, acc[j], false, false);
    }
    int mbase = (tm << 4) + (hi ? 8 : 0);    // C/D: vgpr r -> M=r(+8)
    #pragma unroll
    for (int j = 0; j < NT; ++j) {
        int col = colBase + j * 16 + l15;
        if (col < F) {
            #pragma unroll
            for (int r = 0; r < 8; ++r)
                Y[(size_t)(mbase + r) * F + col] = acc[j][r];
        }
    }
}

// ---- attention scalar products: a_s[n,h], a_d[n,h] ------------------------

__global__ void gat_scores(const float* __restrict__ xh,
                           const float* __restrict__ asr, const float* __restrict__ adr,
                           float* __restrict__ as_, float* __restrict__ ad_,
                           int N, int H, int C) {
    long t = (long)blockIdx.x * blockDim.x + threadIdx.x;
    if (t >= (long)N * H) return;
    int n = (int)(t / H), h = (int)(t % H);
    const float* x  = xh  + (size_t)n * H * C + (size_t)h * C;
    const float* a1 = asr + (size_t)h * C;
    const float* a2 = adr + (size_t)h * C;
    float s1 = 0.f, s2 = 0.f;
    for (int c = 0; c < C; ++c) { s1 += x[c] * a1[c]; s2 += x[c] * a2[c]; }
    as_[t] = s1; ad_[t] = s2;
}

// ---- edge pass 1: segment max (monotone-uint atomicMax) -------------------

__global__ void gat_edge_max(const int* __restrict__ esrc, const int* __restrict__ edst,
                             const float* __restrict__ as_, const float* __restrict__ ad_,
                             unsigned* __restrict__ mx, long E, int N, int H) {
    long t = (long)blockIdx.x * blockDim.x + threadIdx.x;
    long EP = E + N;
    if (t >= EP * H) return;
    long eid = t / H; int h = (int)(t % H);
    int s, d; edge_nodes(eid, esrc, edst, E, s, d);
    float e = as_[(size_t)s * H + h] + ad_[(size_t)d * H + h];
    e = e > 0.f ? e : NEG_SLOPE * e;
    atomicMax(&mx[(size_t)d * H + h], fkey(e));
}

// ---- edge pass 2: alpha = exp(e - m), denom accumulation ------------------

__global__ void gat_edge_alpha(const int* __restrict__ esrc, const int* __restrict__ edst,
                               const float* __restrict__ as_, const float* __restrict__ ad_,
                               const unsigned* __restrict__ mx,
                               float* __restrict__ alpha, float* __restrict__ dn,
                               long E, int N, int H) {
    long t = (long)blockIdx.x * blockDim.x + threadIdx.x;
    long EP = E + N;
    if (t >= EP * H) return;
    long eid = t / H; int h = (int)(t % H);
    int s, d; edge_nodes(eid, esrc, edst, E, s, d);
    float e = as_[(size_t)s * H + h] + ad_[(size_t)d * H + h];
    e = e > 0.f ? e : NEG_SLOPE * e;
    float a = __expf(e - funkey(mx[(size_t)d * H + h]));
    alpha[t] = a;
    atomicAdd(&dn[(size_t)d * H + h], a);
}

// ---- edge pass 3: scatter messages, one wave32 per edge -------------------

__global__ __launch_bounds__(256) void gat_edge_msg(
        const int* __restrict__ esrc, const int* __restrict__ edst,
        const float* __restrict__ alpha, const float* __restrict__ dn,
        const float* __restrict__ xh, float* __restrict__ agg,
        long E, int N, int H, int C, int F) {
    int lane = threadIdx.x & 31;
    long eid = ((long)blockIdx.x * blockDim.x + threadIdx.x) >> 5;
    long EP = E + N;
    if (eid >= EP) return;
    int s, d; edge_nodes(eid, esrc, edst, E, s, d);
    __builtin_prefetch(xh + (size_t)s * F + lane, 0, 0);   // global_prefetch_b8
    for (int c = lane; c < F; c += 32) {
        int h = c / C;
        float w = alpha[eid * H + h] / dn[(size_t)d * H + h];
        atomicAdd(&agg[(size_t)d * F + c], w * xh[(size_t)s * F + c]);
    }
}

// ---- node epilogues -------------------------------------------------------

__global__ void gat_bias_relu(float* __restrict__ agg, const float* __restrict__ b,
                              long total, int F) {
    long t = (long)blockIdx.x * blockDim.x + threadIdx.x;
    if (t >= total) return;
    float v = agg[t] + b[t % F];
    agg[t] = v > 0.f ? v : 0.f;
}

__global__ __launch_bounds__(256) void gat_logsoftmax(
        const float* __restrict__ agg, const float* __restrict__ b,
        float* __restrict__ out, int N, int F) {
    int lane = threadIdx.x & 31;
    long n = ((long)blockIdx.x * blockDim.x + threadIdx.x) >> 5;
    if (n >= N) return;
    int c1 = lane + 32;
    float v0 = (lane < F) ? agg[n * F + lane] + b[lane] : -__builtin_inff();
    float v1 = (c1 < F)   ? agg[n * F + c1]   + b[c1]   : -__builtin_inff();
    float m = fmaxf(v0, v1);
    #pragma unroll
    for (int o = 16; o; o >>= 1) m = fmaxf(m, __shfl_xor(m, o, 32));
    float se = (lane < F ? __expf(v0 - m) : 0.f) + (c1 < F ? __expf(v1 - m) : 0.f);
    #pragma unroll
    for (int o = 16; o; o >>= 1) se += __shfl_xor(se, o, 32);
    float ls = __logf(se);
    if (lane < F) out[n * F + lane] = v0 - m - ls;
    if (c1 < F)   out[n * F + c1]   = v1 - m - ls;
}

// ---------------------------------------------------------------------------

static inline unsigned cdiv(long a, long b) { return (unsigned)((a + b - 1) / b); }

extern "C" void kernel_launch(void* const* d_in, const int* in_sizes, int n_in,
                              void* d_out, int out_size, void* d_ws, size_t ws_size,
                              hipStream_t stream) {
    const int*   edge    = (const int*)d_in[0];
    const float* feat    = (const float*)d_in[1];
    const float* W1      = (const float*)d_in[2];
    const float* a_src1  = (const float*)d_in[3];
    const float* a_dst1  = (const float*)d_in[4];
    const float* b1      = (const float*)d_in[5];
    const float* W2      = (const float*)d_in[6];
    const float* a_src2  = (const float*)d_in[7];
    const float* a_dst2  = (const float*)d_in[8];
    const float* b2      = (const float*)d_in[9];
    const float* W3      = (const float*)d_in[10];
    const float* a_src3  = (const float*)d_in[11];
    const float* a_dst3  = (const float*)d_in[12];
    const float* b3      = (const float*)d_in[13];

    const long E = in_sizes[0] / 2;
    const int  N = in_sizes[1] / 256;
    const long EP = E + (long)N;
    const int  Hmax = 4;
    const int* esrc = edge;
    const int* edst = edge + E;

    // workspace carve-out (256B aligned)
    size_t off = 0;
    char* ws = (char*)d_ws;
    auto carve = [&](size_t bytes) -> char* {
        char* p = ws + off;
        off = (off + bytes + 255) & ~(size_t)255;
        return p;
    };
    unsigned short* xb   = (unsigned short*)carve((size_t)N * 256 * 2); // bf16 input
    unsigned short* wtb  = (unsigned short*)carve(128 * 256 * 2);       // bf16 W^T (padded)
    float*    xh    = (float*)carve((size_t)N * 128 * 4);               // X@W
    float*    agg   = (float*)carve((size_t)N * 128 * 4);               // aggregate
    float*    as_   = (float*)carve((size_t)N * Hmax * 4);
    float*    ad_   = (float*)carve((size_t)N * Hmax * 4);
    unsigned* mx    = (unsigned*)carve((size_t)N * Hmax * 4);
    float*    dn    = (float*)carve((size_t)N * Hmax * 4);
    float*    alpha = (float*)carve((size_t)EP * Hmax * 4);
    (void)ws_size; (void)n_in; (void)out_size;

    const int T = 256;
    auto run_layer = [&](const float* W, const float* asr, const float* adr,
                         int K, int F, int H) {
        int C = F / H;
        int colBlocks = (F + 63) >> 6;
        int Fpad = colBlocks * 64;
        unsigned gemmWaves = (unsigned)(N / 16) * (unsigned)colBlocks;
        unsigned gemmBlocks = cdiv((long)gemmWaves * 32, T);
        // zero-pad Wt rows F..Fpad-1 so the GEMM K-loop is branchless
        gat_zero       <<<cdiv((long)Fpad * K / 2, T), T, 0, stream>>>((unsigned*)wtb, (long)Fpad * K / 2);
        gat_cvt_wt     <<<cdiv((long)K * F, T), T, 0, stream>>>(W, wtb, K, F);
        if (K == 256)
            gat_wmma_gemm<256><<<gemmBlocks, T, 0, stream>>>(xb, wtb, xh, N, F);
        else
            gat_wmma_gemm<128><<<gemmBlocks, T, 0, stream>>>(xb, wtb, xh, N, F);
        gat_scores     <<<cdiv((long)N * H, T), T, 0, stream>>>(xh, asr, adr, as_, ad_, N, H, C);
        gat_zero       <<<cdiv((long)N * H, T), T, 0, stream>>>(mx, (long)N * H);
        gat_zero       <<<cdiv((long)N * H, T), T, 0, stream>>>((unsigned*)dn, (long)N * H);
        gat_zero       <<<cdiv((long)N * F, T), T, 0, stream>>>((unsigned*)agg, (long)N * F);
        gat_edge_max   <<<cdiv(EP * H, T), T, 0, stream>>>(esrc, edst, as_, ad_, mx, E, N, H);
        gat_edge_alpha <<<cdiv(EP * H, T), T, 0, stream>>>(esrc, edst, as_, ad_, mx, alpha, dn, E, N, H);
        gat_edge_msg   <<<cdiv(EP * 32, T), T, 0, stream>>>(esrc, edst, alpha, dn, xh, agg, E, N, H, C, F);
    };

    // ---- layer 1: 256 -> 128, 4 heads, relu ----
    gat_cvt_bf16<<<cdiv((long)N * 256, T), T, 0, stream>>>(feat, xb, (long)N * 256);
    run_layer(W1, a_src1, a_dst1, 256, 128, 4);
    gat_bias_relu<<<cdiv((long)N * 128, T), T, 0, stream>>>(agg, b1, (long)N * 128, 128);

    // ---- layer 2: 128 -> 128, 1 head, relu ----
    gat_cvt_bf16<<<cdiv((long)N * 128, T), T, 0, stream>>>(agg, xb, (long)N * 128);
    run_layer(W2, a_src2, a_dst2, 128, 128, 1);
    gat_bias_relu<<<cdiv((long)N * 128, T), T, 0, stream>>>(agg, b2, (long)N * 128, 128);

    // ---- layer 3: 128 -> 40, 1 head, log_softmax ----
    gat_cvt_bf16<<<cdiv((long)N * 128, T), T, 0, stream>>>(agg, xb, (long)N * 128);
    run_layer(W3, a_src3, a_dst3, 128, 40, 1);
    gat_logsoftmax<<<cdiv((long)N * 32, T), T, 0, stream>>>(agg, b3, (float*)d_out, N, 40);
}
